// Nucleus1TemporalAttention_1829656068447
// MI455X (gfx1250) — matrix operand order repaired and checked by reading, not compiled, not run on hardware
//
#include <hip/hip_runtime.h>

// ---------------------------------------------------------------------------
// Types / helpers
// ---------------------------------------------------------------------------
typedef unsigned short u16;
typedef unsigned int   u32;
typedef __bf16 bf16;
typedef bf16  v16bf __attribute__((ext_vector_type(16)));
typedef float v8f   __attribute__((ext_vector_type(8)));
typedef unsigned int u32x4 __attribute__((ext_vector_type(4)));
typedef int          i32x4 __attribute__((ext_vector_type(4)));
typedef int          i32x8 __attribute__((ext_vector_type(8)));

#if defined(__has_builtin)
#  if __has_builtin(__builtin_amdgcn_tensor_load_to_lds) && \
      __has_builtin(__builtin_amdgcn_s_wait_tensorcnt)
#    define HAVE_TDM 1
#  endif
#endif
#ifndef HAVE_TDM
#  define HAVE_TDM 0
#endif

__device__ __forceinline__ float bf2f(u16 h) {
    return __uint_as_float(((u32)h) << 16);
}
__device__ __forceinline__ u16 f2bf(float f) {      // round-to-nearest-even
    u32 u = __float_as_uint(f);
    u32 r = u + 0x7FFFu + ((u >> 16) & 1u);
    return (u16)(r >> 16);
}
__device__ __forceinline__ u32 pack2bf(float lo, float hi) {
    return (u32)f2bf(lo) | ((u32)f2bf(hi) << 16);
}
__device__ __forceinline__ v8f vzero8() {
    v8f z = {0.f, 0.f, 0.f, 0.f, 0.f, 0.f, 0.f, 0.f};
    return z;
}

// ---------------------------------------------------------------------------
// Tensor Data Mover: async DMA of a 2D bf16 tile (tile_k x tile_rows, row
// stride in elements) from global memory into LDS.  D# per ISA 08 §8:
//   group0: count=1 | lds_addr | global_addr[56:0] | type=2
//   group1: data_size=2B, optional pad (4 DWORDs per 16 DWORDs -> LDS row
//           stride 80B for a 64B tile row), tensor dims (huge: no OOB clip),
//           tile_dim0/1, 48-bit tensor_dim0_stride
// Wave-level op (EXEC ignored): completion via s_wait_tensorcnt (in-order).
// ---------------------------------------------------------------------------
__device__ __forceinline__ void tdm_load_2d_bf16(
    u32 lds_off_bytes, const u16* gptr,
    u32 tile_k, u32 tile_rows, unsigned long long row_stride_elems,
    bool pad_16B_per_64B)
{
#if HAVE_TDM
    unsigned long long ga = (unsigned long long)(size_t)gptr;
    u32x4 g0;
    g0.x = 1u;                                           // count=1 (valid)
    g0.y = lds_off_bytes;                                // lds_addr [63:32]
    g0.z = (u32)ga;                                      // global_addr[31:0]
    g0.w = (u32)((ga >> 32) & 0x01FFFFFFu) | (2u << 30); // addr[56:32] | type=2
    u32 d0 = (1u << 16);                                 // data_size = 2 bytes
    if (pad_16B_per_64B)
        d0 |= (1u << 20) | (3u << 22) | (3u << 25);      // pad_en | every 16 dw | 4 dw
    const u32 td0 = 0x7FFFFFFFu, td1 = 0x7FFFFFFFu;      // huge tensor dims
    i32x8 g1;
    g1[0] = (int)d0;
    g1[1] = (int)((td0 & 0xFFFFu) << 16);                // tensor_dim0 lo
    g1[2] = (int)((td0 >> 16) | ((td1 & 0xFFFFu) << 16));
    g1[3] = (int)((td1 >> 16) | (tile_k << 16));         // tile_dim0
    g1[4] = (int)(tile_rows & 0xFFFFu);                  // tile_dim1 (tile_dim2=0)
    g1[5] = (int)(u32)row_stride_elems;                  // stride[31:0]
    g1[6] = (int)(u32)((row_stride_elems >> 32) & 0xFFFFu);
    g1[7] = 0;
    i32x4 gz4 = {0, 0, 0, 0};
#if __clang_major__ >= 23
    i32x8 gz8 = {0, 0, 0, 0, 0, 0, 0, 0};
    __builtin_amdgcn_tensor_load_to_lds(g0, g1, gz4, gz4, gz8, 0);
#else
    __builtin_amdgcn_tensor_load_to_lds(g0, g1, gz4, gz4, 0);
#endif
#else
    (void)lds_off_bytes; (void)gptr; (void)tile_k; (void)tile_rows;
    (void)row_stride_elems; (void)pad_16B_per_64B;
#endif
}

// ---------------------------------------------------------------------------
// WMMA fragment gathers from LDS (ISA 7.12.2 wave32 layouts)
// ---------------------------------------------------------------------------
union FragU { v16bf v; u32 u[8]; };

// A-matrix 16x32 bf16: lane m = lane&15, half = lane>>4;
// element e: K = 16*(e>>3) + 8*half + (e&7)
__device__ __forceinline__ v16bf load_a_frag(const u16* base, int ld, int lane) {
    const int m = lane & 15, half = lane >> 4;
    const u16* row = base + m * ld;
    FragU r;
#pragma unroll
    for (int p = 0; p < 8; ++p) {
        int k = ((p >> 2) << 4) + (half << 3) + ((2 * p) & 7);
        r.u[p] = *(const u32*)(row + k);
    }
    return r.v;
}

// B-matrix 32x16 bf16; LDS tile stored [n][k]: lane n = lane&15, half = lane>>4;
// element e: K = 16*(e>>3) + 8*half + (e&7)
__device__ __forceinline__ v16bf load_b_frag(const u16* base, int ld, int lane) {
    const int n = lane & 15, half = lane >> 4;
    const u16* row = base + n * ld + half * 8;
    FragU r;
#pragma unroll
    for (int p = 0; p < 8; ++p) r.u[p] = *(const u32*)(row + ((p >> 2) << 4) + ((2 * p) & 7));
    return r.v;
}

// ---------------------------------------------------------------------------
// Kernels 1 & 3: tiled WMMA GEMM.  out[m,n] = sum_k A[m,k] * W[n,k] + bias[n]
// Block tile 128(M) x 128(N) x 32(K); 256 threads = 8 waves in 4(M) x 2(N).
// Double-buffered software pipeline: the Tensor Data Mover stages tile i+1
// (TENSORcnt-gated: in-order completion lets us wait "<= ops in flight")
// while the waves run WMMAs on tile i.  fp32 A (TDM cannot convert) is
// register-pipelined: global_load for tile i+1 issues before the WMMAs of
// tile i, converts/stores to LDS afterwards; next-next tile prefetched.
// ---------------------------------------------------------------------------
template <bool A_IS_F32, bool OUT_F32>
__global__ __launch_bounds__(256) void wmma_gemm(
    const void* __restrict__ Araw, const u16* __restrict__ W,
    const u16* __restrict__ bias, void* __restrict__ Out,
    int N, int K)
{
    __shared__ u16 As[2][128][40];  // 80B row stride: 16B-aligned, conflict-spread
    __shared__ u16 Bs[2][128][40];

    const int tid  = threadIdx.x;
    const int lane = tid & 31;
    const int wave = tid >> 5;
    const int wm   = wave >> 1;               // 0..3 -> 32-row strip
    const int wn   = wave & 1;                // 0..1 -> 64-col strip
    const long mbase = (long)blockIdx.y * 128;
    const long nbase = (long)blockIdx.x * 128;
    const int steps = K / 32;

    v8f acc[2][4];
#pragma unroll
    for (int mi = 0; mi < 2; ++mi)
#pragma unroll
        for (int ni = 0; ni < 4; ++ni) acc[mi][ni] = vzero8();

#if HAVE_TDM
    auto issue_tdm = [&](int buf, int k0) {
        tdm_load_2d_bf16((u32)(size_t)&Bs[buf][0][0],
                         W + nbase * (long)K + k0,
                         32u, 128u, (unsigned long long)K, true);
        if constexpr (!A_IS_F32)
            tdm_load_2d_bf16((u32)(size_t)&As[buf][0][0],
                             (const u16*)Araw + mbase * (long)K + k0,
                             32u, 128u, (unsigned long long)K, true);
    };
#else
    auto stage_manual = [&](int buf, int k0) {
#pragma unroll
        for (int j = 0; j < 2; ++j) {
            int id = tid + j * 256;
            int r = id >> 2, c = (id & 3) * 8;
            *(uint4*)&Bs[buf][r][c] =
                *(const uint4*)(W + (nbase + r) * (long)K + k0 + c);
        }
        if constexpr (!A_IS_F32) {
            const u16* A = (const u16*)Araw;
#pragma unroll
            for (int j = 0; j < 2; ++j) {
                int id = tid + j * 256;
                int r = id >> 2, c = (id & 3) * 8;
                *(uint4*)&As[buf][r][c] =
                    *(const uint4*)(A + (mbase + r) * (long)K + k0 + c);
            }
        }
    };
#endif

    // ---- fp32 A register pipeline state ----
    int ar[4], ac[4];
    float4 areg[4];
    if constexpr (A_IS_F32) {
        const float* A = (const float*)Araw;
#pragma unroll
        for (int j = 0; j < 4; ++j) {
            int id = tid + j * 256;
            ar[j] = id >> 3;                  // row in 128-row tile
            ac[j] = (id & 7) * 4;             // col (float4 granule)
            areg[j] = *(const float4*)(A + (mbase + ar[j]) * (long)K + ac[j]);
        }
    }

    // ---- prologue: tile 0 in flight ----
#if HAVE_TDM
    if (wave == 0) issue_tdm(0, 0);
#else
    stage_manual(0, 0);
#endif

    for (int i = 0; i < steps; ++i) {
        const int  cur  = i & 1, nxt = cur ^ 1;
        const int  k0   = i * 32;
        const bool more = (i + 1 < steps);

        if constexpr (A_IS_F32) {
            // commit regs (loaded last iteration) for THIS tile into LDS
#pragma unroll
            for (int j = 0; j < 4; ++j) {
                *(u32*)&As[cur][ar[j]][ac[j]]     = pack2bf(areg[j].x, areg[j].y);
                *(u32*)&As[cur][ar[j]][ac[j] + 2] = pack2bf(areg[j].z, areg[j].w);
            }
            // issue global loads for tile i+1 (latency hides under WMMAs)
            if (more) {
                const float* A = (const float*)Araw;
#pragma unroll
                for (int j = 0; j < 4; ++j)
                    areg[j] = *(const float4*)(A + (mbase + ar[j]) * (long)K +
                                               (k0 + 32) + ac[j]);
                if (i + 2 < steps)            // global_prefetch_b8, 2 tiles ahead
                    __builtin_prefetch(A + (mbase + (tid >> 1)) * (long)K +
                                       (k0 + 64) + (tid & 1) * 16, 0, 1);
            }
        }

#if HAVE_TDM
        if (wave == 0) {
            if (more) {
                issue_tdm(nxt, k0 + 32);      // next tile in flight
                // older batch must complete; newest batch may stay outstanding
                if constexpr (A_IS_F32) __builtin_amdgcn_s_wait_tensorcnt(1);
                else                    __builtin_amdgcn_s_wait_tensorcnt(2);
            } else {
                __builtin_amdgcn_s_wait_tensorcnt(0);
            }
        }
#else
        if (more) stage_manual(nxt, k0 + 32);
#endif
        __syncthreads();

        // ---- WMMA on tile i: each wave 2(M) x 4(N) 16x16 tiles, K=32 ----
#pragma unroll
        for (int mi = 0; mi < 2; ++mi) {
            v16bf a = load_a_frag(&As[cur][wm * 32 + mi * 16][0], 40, lane);
#pragma unroll
            for (int ni = 0; ni < 4; ++ni) {
                v16bf b = load_b_frag(&Bs[cur][wn * 64 + ni * 16][0], 40, lane);
                acc[mi][ni] = __builtin_amdgcn_wmma_f32_16x16x32_bf16(
                    false, a, false, b, (short)0, acc[mi][ni], false, false);
            }
        }
        __syncthreads();
    }

    // ---- epilogue: bias + store (D layout: n = lane&15, M = r + 8*(lane>>4)) ----
    const int half = lane >> 4, nl = lane & 15;
#pragma unroll
    for (int mi = 0; mi < 2; ++mi) {
#pragma unroll
        for (int ni = 0; ni < 4; ++ni) {
            long col = nbase + wn * 64 + ni * 16 + nl;
            float bv = bf2f(bias[col]);
#pragma unroll
            for (int r8 = 0; r8 < 8; ++r8) {
                long row = mbase + wm * 32 + mi * 16 + half * 8 + r8;
                float v = acc[mi][ni][r8] + bv;
                if constexpr (OUT_F32)
                    ((float*)Out)[row * (long)N + col] = v;
                else
                    ((u16*)Out)[row * (long)N + col] = f2bf(v);
            }
        }
    }
}

// ---------------------------------------------------------------------------
// Kernel 2: fused LayerNorm + RoPE + attention, one wave32 per head.
// T=32 (= wave32: one row per lane), HD=64, ROT=32.  The head's full q|k|v
// block (32 rows x 192 bf16, row stride 3072*1024 elems) is fetched by ONE
// TDM descriptor while the wave computes its RoPE tables.
// ---------------------------------------------------------------------------
__global__ __launch_bounds__(32) void attn_kernel(
    const u16* __restrict__ qkv, u16* __restrict__ o,
    const u16* __restrict__ qg, const u16* __restrict__ qb,
    const u16* __restrict__ kg, const u16* __restrict__ kb)
{
#if HAVE_TDM
    __shared__ u16  qkvs[32][192];  // raw q|k|v rows for this head (12 KB)
#endif
    __shared__ u16  qs[32][68];     // q after LN+RoPE, [t][hd]
    __shared__ u16  ks[32][68];     // k after LN+RoPE, [t][hd]
    __shared__ u16  vT[64][36];     // v transposed, [hd][t]
    __shared__ float Ss[32][32];    // fp32 scores
    __shared__ u16  att[32][36];    // softmax(scores) bf16
    __shared__ u16  os[32][68];     // output tile [t][hd]
    __shared__ u16  gq[64], bq[64], gk[64], bk[64];

    const int lane = threadIdx.x;
    const int hid  = blockIdx.x;            // ((b*H + h)*W + w)*16 + nh
    const int nh   = hid & 15;
    const int n    = hid >> 4;
    const int w    = n & 31;
    const int h    = (n >> 5) & 31;
    const int b    = n >> 10;

    const int  t    = lane;                                      // row per lane
    const long tok0 = ((long)(b * 32)) * 1024 + h * 32 + w;      // token at t=0
    const long tok  = tok0 + (long)t * 1024;                     // H*W = 1024

#if HAVE_TDM
    // one DMA for the whole head: 32 rows x 192 bf16, row stride 6 MB
    tdm_load_2d_bf16((u32)(size_t)&qkvs[0][0],
                     qkv + tok0 * 3072 + (long)nh * 192,
                     192u, 32u, 3145728ULL /* 1024*3072 */, false);
#endif

    gq[lane] = qg[lane]; gq[lane + 32] = qg[lane + 32];
    bq[lane] = qb[lane]; bq[lane + 32] = qb[lane + 32];
    gk[lane] = kg[lane]; gk[lane + 32] = kg[lane + 32];
    bk[lane] = kb[lane]; bk[lane + 32] = kb[lane + 32];

    // RoPE tables for this row (overlaps the TDM):  inv_freq[i] = 10000^(-2i/32)
    float cosv[16], sinv[16];
#pragma unroll
    for (int i = 0; i < 16; ++i) {
        float inv = exp2f(-(float)(2 * i) * (13.287712379549449f / 32.0f));
        float ang = (float)t * inv;
        cosv[i] = __cosf(ang);
        sinv[i] = __sinf(ang);
    }

#if HAVE_TDM
    __builtin_amdgcn_s_wait_tensorcnt(0);
    const u16* rowq = &qkvs[t][0];
#else
    const u16* rowq = qkv + tok * 3072 + (long)nh * 192;
#endif
    __syncthreads();   // single-wave WG: barrier is free

    // ---- Q and K: load row, layernorm, rope, store to LDS ----
#pragma unroll
    for (int qk = 0; qk < 2; ++qk) {
        const u16* row = rowq + qk * 64;
        const u16* g  = qk ? gk : gq;
        const u16* be = qk ? bk : bq;
        float f[64];
#pragma unroll
        for (int i = 0; i < 8; ++i) {
            uint4 d = ((const uint4*)row)[i];
            u32 dd[4] = {d.x, d.y, d.z, d.w};
#pragma unroll
            for (int j = 0; j < 4; ++j) {
                f[i * 8 + j * 2]     = bf2f((u16)(dd[j] & 0xFFFFu));
                f[i * 8 + j * 2 + 1] = bf2f((u16)(dd[j] >> 16));
            }
        }
        float m = 0.f;
#pragma unroll
        for (int i = 0; i < 64; ++i) m += f[i];
        m *= (1.0f / 64.0f);
        float var = 0.f;
#pragma unroll
        for (int i = 0; i < 64; ++i) { float d = f[i] - m; var += d * d; }
        var *= (1.0f / 64.0f);
        float rs = rsqrtf(var + 1e-5f);
#pragma unroll
        for (int i = 0; i < 64; ++i)
            f[i] = (f[i] - m) * rs * bf2f(g[i]) + bf2f(be[i]);
        // RoPE on first 32 dims
#pragma unroll
        for (int i = 0; i < 16; ++i) {
            float x0 = f[2 * i], x1 = f[2 * i + 1];
            f[2 * i]     = x0 * cosv[i] - x1 * sinv[i];
            f[2 * i + 1] = x1 * cosv[i] + x0 * sinv[i];
        }
        u16 (*dst)[68] = qk ? ks : qs;
#pragma unroll
        for (int i = 0; i < 32; ++i)
            *(u32*)&dst[t][2 * i] = pack2bf(f[2 * i], f[2 * i + 1]);
    }

    // ---- V: copy row, store transposed [hd][t] ----
    {
        const u16* rowv = rowq + 128;
#pragma unroll
        for (int i = 0; i < 8; ++i) {
            uint4 d = ((const uint4*)rowv)[i];
            u32 dd[4] = {d.x, d.y, d.z, d.w};
#pragma unroll
            for (int j = 0; j < 4; ++j) {
                vT[i * 8 + j * 2][t]     = (u16)(dd[j] & 0xFFFFu);
                vT[i * 8 + j * 2 + 1][t] = (u16)(dd[j] >> 16);
            }
        }
    }
    __syncthreads();

    // ---- scores = Q(32x64) @ K^T(64x32) * 0.125 : 4 tiles, 2 K-steps each ----
    const int half = lane >> 4, nl = lane & 15;
#pragma unroll
    for (int mi = 0; mi < 2; ++mi) {
#pragma unroll
        for (int ni = 0; ni < 2; ++ni) {
            v8f acc = vzero8();
#pragma unroll
            for (int kk = 0; kk < 2; ++kk) {
                v16bf a    = load_a_frag(&qs[mi * 16][kk * 32], 68, lane);
                v16bf bmat = load_b_frag(&ks[ni * 16][kk * 32], 68, lane);
                acc = __builtin_amdgcn_wmma_f32_16x16x32_bf16(
                    false, a, false, bmat, (short)0, acc, false, false);
            }
#pragma unroll
            for (int r8 = 0; r8 < 8; ++r8)
                Ss[mi * 16 + half * 8 + r8][ni * 16 + nl] = acc[r8] * 0.125f;
        }
    }
    __syncthreads();

    // ---- fp32 softmax, row per lane ----
    {
        float s[32];
        float mx = -3.4e38f;
#pragma unroll
        for (int c = 0; c < 32; ++c) { s[c] = Ss[t][c]; mx = fmaxf(mx, s[c]); }
        float sum = 0.f;
#pragma unroll
        for (int c = 0; c < 32; ++c) { s[c] = __expf(s[c] - mx); sum += s[c]; }
        float inv = 1.0f / sum;
#pragma unroll
        for (int c = 0; c < 16; ++c)
            *(u32*)&att[t][2 * c] = pack2bf(s[2 * c] * inv, s[2 * c + 1] * inv);
    }
    __syncthreads();

    // ---- O = attn(32x32) @ V(32x64): 8 tiles, K=32 -> one WMMA each ----
#pragma unroll
    for (int mi = 0; mi < 2; ++mi) {
        v16bf a = load_a_frag(&att[mi * 16][0], 36, lane);
#pragma unroll
        for (int ni = 0; ni < 4; ++ni) {
            v16bf bmat = load_b_frag(&vT[ni * 16][0], 36, lane);
            v8f acc = vzero8();
            acc = __builtin_amdgcn_wmma_f32_16x16x32_bf16(
                false, a, false, bmat, (short)0, acc, false, false);
#pragma unroll
            for (int r8 = 0; r8 < 8; ++r8)
                os[mi * 16 + half * 8 + r8][ni * 16 + nl] = f2bf(acc[r8]);
        }
    }
    __syncthreads();

    // ---- store O row (coalesced 16B stores) ----
    u16* orow = o + tok * 1024 + (long)nh * 64;
#pragma unroll
    for (int i = 0; i < 8; ++i) {
        uint4 d;
        d.x = *(u32*)&os[t][i * 8 + 0];
        d.y = *(u32*)&os[t][i * 8 + 2];
        d.z = *(u32*)&os[t][i * 8 + 4];
        d.w = *(u32*)&os[t][i * 8 + 6];
        ((uint4*)orow)[i] = d;
    }
}

// ---------------------------------------------------------------------------
// Host launch
// ---------------------------------------------------------------------------
extern "C" void kernel_launch(void* const* d_in, const int* in_sizes, int n_in,
                              void* d_out, int out_size, void* d_ws, size_t ws_size,
                              hipStream_t stream) {
    (void)in_sizes; (void)n_in; (void)out_size; (void)ws_size;
    const float* x     = (const float*)d_in[0];
    const u16* w_qkv   = (const u16*)d_in[1];
    const u16* b_qkv   = (const u16*)d_in[2];
    const u16* q_gamma = (const u16*)d_in[3];
    const u16* q_beta  = (const u16*)d_in[4];
    const u16* k_gamma = (const u16*)d_in[5];
    const u16* k_beta  = (const u16*)d_in[6];
    const u16* w_out   = (const u16*)d_in[7];
    const u16* b_out   = (const u16*)d_in[8];

    const long M = 65536;                 // B*T*H*W tokens
    u16* qkv_ws = (u16*)d_ws;                         // 65536 x 3072 bf16 (384 MB)
    u16* o_ws   = qkv_ws + M * 3072;                  // 65536 x 1024 bf16 (128 MB)

    // 1) QKV projection: fp32 x -> bf16, N=3072, K=1024
    wmma_gemm<true, false><<<dim3(3072 / 128, M / 128), 256, 0, stream>>>(
        (const void*)x, w_qkv, b_qkv, (void*)qkv_ws, 3072, 1024);

    // 2) Fused LN + RoPE + attention: one wave per (b,h,w,nh) head
    attn_kernel<<<dim3(32768), 32, 0, stream>>>(
        qkv_ws, o_ws, q_gamma, q_beta, k_gamma, k_beta);

    // 3) Output projection: bf16 o -> fp32 out, N=1024, K=1024
    wmma_gemm<false, true><<<dim3(1024 / 128, M / 128), 256, 0, stream>>>(
        (const void*)o_ws, w_out, b_out, d_out, 1024, 1024);
}
